// Block_24541443129820
// MI455X (gfx1250) — matrix operand order repaired and checked by reading, not compiled, not run on hardware
//
#include <hip/hip_runtime.h>
#include <hip/hip_bf16.h>
#include <math.h>

// ---------------- problem constants ----------------
#define NB   8
#define NT_  1024
#define ND   768
#define NH   12
#define NDH  64
#define NE   16
#define NDFF 3072
#define NTOK (NB * NT_)
#define LNEPS 1e-5f

typedef __bf16 bf16;
typedef __attribute__((ext_vector_type(16))) __bf16 v16bf;
typedef __attribute__((ext_vector_type(8)))  __bf16 v8bf;
typedef __attribute__((ext_vector_type(8)))  float  v8f;

__device__ __forceinline__ bf16 f2b(float f) { return (bf16)f; }

__device__ __forceinline__ v8f wmma_bf16(v16bf a, v16bf b, v8f c) {
  // D = A(16x32 bf16) * B(32x16 bf16) + C(16x16 f32)
  return __builtin_amdgcn_wmma_f32_16x16x32_bf16(
      /*neg_a=*/false, a, /*neg_b=*/false, b,
      /*c_mod=*/(short)0, c, /*reuse_a=*/false, /*reuse_b=*/false);
}

// A fragment: lane L holds row m0+(L&15); K chunks [kb,kb+8) and [kb+16,kb+24),
// kb = k0 + (L>=16 ? 8 : 0).  A is row-major [M][lda] bf16, 16B-aligned rows.
__device__ __forceinline__ v16bf load_a_frag(const bf16* __restrict__ A, int lda,
                                             int m0, int k0, int lane) {
  int row = m0 + (lane & 15);
  int kb  = k0 + ((lane & 16) ? 8 : 0);
  const bf16* p = A + (size_t)row * lda + kb;
  v8bf lo = *(const v8bf*)(p);
  v8bf hi = *(const v8bf*)(p + 16);
  v16bf a;
#pragma unroll
  for (int i = 0; i < 8; ++i) { a[i] = lo[i]; a[i + 8] = hi[i]; }
  return a;
}

// B fragment from TRANSPOSED weights Bt row-major [N][ldk] bf16:
// lane L holds column n0+(L&15); K chunks [kb,kb+8) and [kb+16,kb+24), kb = k0 + (L>=16 ? 8 : 0).
__device__ __forceinline__ v16bf load_b_frag(const bf16* __restrict__ Bt, int ldk,
                                             int n0, int k0, int lane) {
  int col = n0 + (lane & 15);
  int kb  = k0 + ((lane & 16) ? 8 : 0);
  const bf16* p = Bt + (size_t)col * ldk + kb;
  v8bf lo = *(const v8bf*)(p);
  v8bf hi = *(const v8bf*)(p + 16);
  v16bf b;
#pragma unroll
  for (int i = 0; i < 8; ++i) { b[i] = lo[i]; b[i + 8] = hi[i]; }
  return b;
}

// ---------------- weight convert: fp32 [K][N] -> bf16 [N][K] ----------------
__global__ __launch_bounds__(256) void wconv_kernel(const float* __restrict__ W,
                                                    bf16* __restrict__ Wt,
                                                    int K, int N) {
  const float* Ws = W + (size_t)blockIdx.z * K * N;
  bf16*        Wd = Wt + (size_t)blockIdx.z * K * N;
  __shared__ float tile[16][17];
  int tx = threadIdx.x & 15, ty = threadIdx.x >> 4;
  int kb = blockIdx.y * 16, nb = blockIdx.x * 16;
  tile[ty][tx] = Ws[(size_t)(kb + ty) * N + nb + tx];
  __syncthreads();
  Wd[(size_t)(nb + ty) * K + kb + tx] = f2b(tile[tx][ty]);
}

// ---------------- LayerNorm over last dim (768); one block per token --------
__device__ __forceinline__ float block_sum256(float v, float* sm) {
#pragma unroll
  for (int m = 16; m > 0; m >>= 1) v += __shfl_xor(v, m, 32);
  if ((threadIdx.x & 31) == 0) sm[threadIdx.x >> 5] = v;
  __syncthreads();
  float tot = 0.f;
#pragma unroll
  for (int i = 0; i < 8; ++i) tot += sm[i];
  __syncthreads();
  return tot;
}

__global__ __launch_bounds__(256) void ln_kernel(const float* __restrict__ x,
                                                 const float* __restrict__ w,
                                                 const float* __restrict__ bb,
                                                 float* __restrict__ outf,
                                                 bf16* __restrict__ outb) {
  __shared__ float sm[8];
  int tok = blockIdx.x;
  const float* xr = x + (size_t)tok * ND;
  int tid = threadIdx.x;
  float v[3]; float s = 0.f;
#pragma unroll
  for (int i = 0; i < 3; ++i) { v[i] = xr[tid + i * 256]; s += v[i]; }
  float mean = block_sum256(s, sm) * (1.0f / ND);
  float s2 = 0.f;
#pragma unroll
  for (int i = 0; i < 3; ++i) { float d = v[i] - mean; s2 += d * d; }
  float var = block_sum256(s2, sm) * (1.0f / ND);
  float rstd = rsqrtf(var + LNEPS);
#pragma unroll
  for (int i = 0; i < 3; ++i) {
    int col = tid + i * 256;
    float val = (v[i] - mean) * rstd * w[col] + bb[col];
    if (outf) outf[(size_t)tok * ND + col] = val;
    outb[(size_t)tok * ND + col] = f2b(val);
  }
}

// ---------------- bf16 WMMA GEMM: C[M,N] = A[M,K] @ Bt[N,K]^T + bias --------
enum { EPI_STORE_F32 = 0, EPI_ADD_RESID = 1, EPI_GELU_BF16 = 2, EPI_MOE_ACC = 3 };

template <int EPI>
__global__ __launch_bounds__(256) void gemm_bf16_kernel(
    const bf16* __restrict__ A, const bf16* __restrict__ Bt,
    const float* __restrict__ bias, float* __restrict__ C,
    bf16* __restrict__ Cbf, const float* __restrict__ resid,
    const float* __restrict__ gate, int expert, int M, int N, int K) {
  int lane = threadIdx.x & 31;
  int wid  = threadIdx.x >> 5;
  int m0 = blockIdx.y * 64  + (wid & 1) * 32;
  int n0 = blockIdx.x * 128 + (wid >> 1) * 32;

  v8f acc[2][2] = {};
  for (int k0 = 0; k0 < K; k0 += 32) {
    v16bf a0 = load_a_frag(A, K, m0,      k0, lane);
    v16bf a1 = load_a_frag(A, K, m0 + 16, k0, lane);
    v16bf b0 = load_b_frag(Bt, K, n0,      k0, lane);
    v16bf b1 = load_b_frag(Bt, K, n0 + 16, k0, lane);
    if (k0 + 64 < K) {  // two-tiles-ahead L2 prefetch (global_prefetch_b8)
      __builtin_prefetch(A  + (size_t)(m0 + (lane & 15)) * K + k0 + 64, 0, 1);
      __builtin_prefetch(Bt + (size_t)(n0 + (lane & 15)) * K + k0 + 64, 0, 1);
    }
    acc[0][0] = wmma_bf16(a0, b0, acc[0][0]);
    acc[0][1] = wmma_bf16(a0, b1, acc[0][1]);
    acc[1][0] = wmma_bf16(a1, b0, acc[1][0]);
    acc[1][1] = wmma_bf16(a1, b1, acc[1][1]);
  }

  int colb = lane & 15;
  int rb   = (lane & 16) ? 8 : 0;
#pragma unroll
  for (int i = 0; i < 2; ++i)
#pragma unroll
    for (int j = 0; j < 2; ++j)
#pragma unroll
      for (int r = 0; r < 8; ++r) {
        int row = m0 + i * 16 + rb + r;
        int col = n0 + j * 16 + colb;
        float v = acc[i][j][r] + (bias ? bias[col] : 0.f);
        size_t idx = (size_t)row * N + col;
        if constexpr (EPI == EPI_STORE_F32) {
          C[idx] = v;
        } else if constexpr (EPI == EPI_ADD_RESID) {
          C[idx] = v + resid[idx];
        } else if constexpr (EPI == EPI_GELU_BF16) {
          float g = 0.5f * v * (1.f + erff(v * 0.70710678118654752f));
          Cbf[idx] = f2b(g);
        } else {  // EPI_MOE_ACC
          C[idx] += gate[(size_t)row * NE + expert] * v;
        }
      }
}

// ---------------- qkv pack: fp32 [tok][3*D] -> q/k bf16 [B,H,T,DH], v^T [B,H,DH,T]
__global__ __launch_bounds__(256) void pack_qkv_kernel(const float* __restrict__ qkv,
                                                       bf16* __restrict__ q,
                                                       bf16* __restrict__ k,
                                                       bf16* __restrict__ vt) {
  size_t idx = (size_t)blockIdx.x * 256 + threadIdx.x;  // over NTOK*ND
  int dh = (int)(idx % NDH);
  int h  = (int)((idx / NDH) % NH);
  size_t tok = idx / ND;
  int t = (int)(tok % NT_);
  int b = (int)(tok / NT_);
  const float* base = qkv + tok * (3 * ND) + h * NDH + dh;
  size_t bh = (size_t)b * NH + h;
  q[(bh * NT_ + t) * NDH + dh] = f2b(base[0]);
  k[(bh * NT_ + t) * NDH + dh] = f2b(base[ND]);
  vt[(bh * NDH + dh) * NT_ + t] = f2b(base[2 * ND]);
}

// ---------------- flash attention: one wave = 16 query rows -----------------
__global__ __launch_bounds__(128) void attn_kernel(const bf16* __restrict__ qb,
                                                   const bf16* __restrict__ kb,
                                                   const bf16* __restrict__ vtb,
                                                   float* __restrict__ y) {
  int bh   = blockIdx.y;           // b*H + h
  int wid  = threadIdx.x >> 5;
  int lane = threadIdx.x & 31;
  int q0   = blockIdx.x * 64 + wid * 16;

  const bf16* Q  = qb  + (size_t)bh * NT_ * NDH;
  const bf16* Km = kb  + (size_t)bh * NT_ * NDH;
  const bf16* Vt = vtb + (size_t)bh * NDH * NT_;

  v16bf qf0 = load_a_frag(Q, NDH, q0, 0,  lane);
  v16bf qf1 = load_a_frag(Q, NDH, q0, 32, lane);

  v8f o[4] = {};
  float rmax[8], rsum[8];
#pragma unroll
  for (int r = 0; r < 8; ++r) { rmax[r] = -1e30f; rsum[r] = 0.f; }

  __shared__ __align__(16) bf16 Pl[4][16][32];
  int colb = lane & 15;
  int rb   = (lane & 16) ? 8 : 0;

  for (int kt = 0; kt < q0 + 16; kt += 32) {
    v8f s0 = {}, s1 = {};
    s0 = wmma_bf16(qf0, load_b_frag(Km, NDH, kt,      0,  lane), s0);
    s0 = wmma_bf16(qf1, load_b_frag(Km, NDH, kt,      32, lane), s0);
    s1 = wmma_bf16(qf0, load_b_frag(Km, NDH, kt + 16, 0,  lane), s1);
    s1 = wmma_bf16(qf1, load_b_frag(Km, NDH, kt + 16, 32, lane), s1);

#pragma unroll
    for (int r = 0; r < 8; ++r) {
      int qi = q0 + rb + r;
      float v0 = s0[r] * 0.125f; if (kt + colb      > qi) v0 = -1e30f;
      float v1 = s1[r] * 0.125f; if (kt + 16 + colb > qi) v1 = -1e30f;
      float tm = fmaxf(v0, v1);
#pragma unroll
      for (int m = 8; m > 0; m >>= 1) tm = fmaxf(tm, __shfl_xor(tm, m, 32));
      float nm   = fmaxf(rmax[r], tm);
      float corr = __expf(rmax[r] - nm);
      float e0 = __expf(v0 - nm), e1 = __expf(v1 - nm);
      float ts = e0 + e1;
#pragma unroll
      for (int m = 8; m > 0; m >>= 1) ts += __shfl_xor(ts, m, 32);
      rsum[r] = rsum[r] * corr + ts;
      rmax[r] = nm;
#pragma unroll
      for (int d = 0; d < 4; ++d) o[d][r] *= corr;
      Pl[wid][rb + r][colb]      = f2b(e0);
      Pl[wid][rb + r][colb + 16] = f2b(e1);
    }
    asm volatile("s_wait_dscnt 0x0" ::: "memory");  // wave-local LDS RAW fence
    v16bf pf = load_a_frag(&Pl[wid][0][0], 32, 0, 0, lane);
#pragma unroll
    for (int d = 0; d < 4; ++d)
      o[d] = wmma_bf16(pf, load_b_frag(Vt, NT_, d * 16, kt, lane), o[d]);
  }

  int b_ = bh / NH, h_ = bh % NH;
#pragma unroll
  for (int d = 0; d < 4; ++d)
#pragma unroll
    for (int r = 0; r < 8; ++r) {
      int qi = q0 + rb + r;
      y[((size_t)(b_ * NT_ + qi)) * ND + h_ * NDH + d * 16 + colb] =
          o[d][r] / rsum[r];
    }
}

// ---------------- router softmax + top-2 gate; one wave per token -----------
__global__ __launch_bounds__(256) void router_gate_kernel(const float* __restrict__ z,
                                                          const float* __restrict__ rw,
                                                          float* __restrict__ gate) {
  int tok  = blockIdx.x * 8 + (threadIdx.x >> 5);
  int lane = threadIdx.x & 31;
  const float* zr = z + (size_t)tok * ND;
  float zz[24];
#pragma unroll
  for (int i = 0; i < 24; ++i) zz[i] = zr[lane + i * 32];
  float logits[NE];
#pragma unroll
  for (int e = 0; e < NE; ++e) {
    float s = 0.f;
#pragma unroll
    for (int i = 0; i < 24; ++i) s += zz[i] * rw[(size_t)(lane + i * 32) * NE + e];
#pragma unroll
    for (int m = 16; m > 0; m >>= 1) s += __shfl_xor(s, m, 32);
    logits[e] = s;
  }
  float mx = logits[0];
#pragma unroll
  for (int e = 1; e < NE; ++e) mx = fmaxf(mx, logits[e]);
  float pe[NE]; float se = 0.f;
#pragma unroll
  for (int e = 0; e < NE; ++e) { pe[e] = __expf(logits[e] - mx); se += pe[e]; }
  float inv = 1.f / se;
  int i1 = 0; float p1v = -1.f;
#pragma unroll
  for (int e = 0; e < NE; ++e) if (pe[e] > p1v) { p1v = pe[e]; i1 = e; }
  int i2 = -1; float p2v = -1.f;
#pragma unroll
  for (int e = 0; e < NE; ++e) if (e != i1 && pe[e] > p2v) { p2v = pe[e]; i2 = e; }
  if (lane < NE) {
    float g = (lane == i1) ? p1v * inv : (lane == i2) ? p2v * inv : 0.f;
    gate[(size_t)tok * NE + lane] = g;
  }
}

// ---------------- misc elementwise -----------------------------------------
__global__ __launch_bounds__(256) void f2bf_kernel(const float* __restrict__ in,
                                                   bf16* __restrict__ out) {
  size_t i = (size_t)blockIdx.x * 256 + threadIdx.x;
  out[i] = f2b(in[i]);
}
__global__ __launch_bounds__(256) void add2_kernel(const float* __restrict__ a,
                                                   const float* __restrict__ b,
                                                   float* __restrict__ out) {
  size_t i = (size_t)blockIdx.x * 256 + threadIdx.x;
  out[i] = a[i] + b[i];
}

// ---------------- host orchestration ----------------------------------------
extern "C" void kernel_launch(void* const* d_in, const int* in_sizes, int n_in,
                              void* d_out, int out_size, void* d_ws, size_t ws_size,
                              hipStream_t stream) {
  const float* x       = (const float*)d_in[0];
  const float* ln1_w   = (const float*)d_in[1];
  const float* ln1_b   = (const float*)d_in[2];
  const float* qkv_w   = (const float*)d_in[3];
  const float* qkv_b   = (const float*)d_in[4];
  const float* proj_w  = (const float*)d_in[5];
  const float* proj_b  = (const float*)d_in[6];
  const float* ln2_w   = (const float*)d_in[7];
  const float* ln2_b   = (const float*)d_in[8];
  const float* rw      = (const float*)d_in[9];
  const float* w1      = (const float*)d_in[10];
  const float* b1      = (const float*)d_in[11];
  const float* w2      = (const float*)d_in[12];
  const float* b2      = (const float*)d_in[13];
  float* out = (float*)d_out;

  char* p = (char*)d_ws;
  auto alloc = [&](size_t bytes) -> void* {
    void* r = (void*)p;
    p += (bytes + 255) & ~(size_t)255;
    return r;
  };
  bf16*  qkvwt  = (bf16*)alloc((size_t)ND * 3 * ND * 2);
  bf16*  projwt = (bf16*)alloc((size_t)ND * ND * 2);
  bf16*  w1t    = (bf16*)alloc((size_t)NE * ND * NDFF * 2);
  bf16*  w2t    = (bf16*)alloc((size_t)NE * NDFF * ND * 2);
  bf16*  hbf    = (bf16*)alloc((size_t)NTOK * ND * 2);
  float* qkvf   = (float*)alloc((size_t)NTOK * 3 * ND * 4);
  bf16*  qbf    = (bf16*)alloc((size_t)NTOK * ND * 2);
  bf16*  kbf    = (bf16*)alloc((size_t)NTOK * ND * 2);
  bf16*  vtbf   = (bf16*)alloc((size_t)NTOK * ND * 2);
  float* yf     = (float*)alloc((size_t)NTOK * ND * 4);
  bf16*  ybf    = (bf16*)alloc((size_t)NTOK * ND * 2);
  float* x1     = (float*)alloc((size_t)NTOK * ND * 4);
  float* zf     = (float*)alloc((size_t)NTOK * ND * 4);
  bf16*  zbf    = (bf16*)alloc((size_t)NTOK * ND * 2);
  float* gate   = (float*)alloc((size_t)NTOK * NE * 4);
  bf16*  t1bf   = (bf16*)qkvf;  // alias: qkv fp32 dead after pack (50.3MB <= 75.5MB)

  const size_t NELT = (size_t)NTOK * ND;           // 6,291,456 (divisible by 256)
  const int    EW_BLOCKS = (int)(NELT / 256);

  // 1) one-time weight converts (fp32 [K][N] -> bf16 [N][K])
  wconv_kernel<<<dim3(3 * ND / 16, ND / 16, 1), 256, 0, stream>>>(qkv_w, qkvwt, ND, 3 * ND);
  wconv_kernel<<<dim3(ND / 16, ND / 16, 1), 256, 0, stream>>>(proj_w, projwt, ND, ND);
  wconv_kernel<<<dim3(NDFF / 16, ND / 16, NE), 256, 0, stream>>>(w1, w1t, ND, NDFF);
  wconv_kernel<<<dim3(ND / 16, NDFF / 16, NE), 256, 0, stream>>>(w2, w2t, NDFF, ND);

  // 2) LN1 -> h (bf16 only)
  ln_kernel<<<NTOK, 256, 0, stream>>>(x, ln1_w, ln1_b, nullptr, hbf);

  // 3) qkv = h @ qkv_w + qkv_b   [8192 x 2304]
  gemm_bf16_kernel<EPI_STORE_F32><<<dim3(3 * ND / 128, NTOK / 64), 256, 0, stream>>>(
      hbf, qkvwt, qkv_b, qkvf, nullptr, nullptr, nullptr, 0, NTOK, 3 * ND, ND);

  // 4) pack q/k row-major per head, v transposed per head
  pack_qkv_kernel<<<EW_BLOCKS, 256, 0, stream>>>(qkvf, qbf, kbf, vtbf);

  // 5) causal flash attention -> y [B,T,D] fp32
  attn_kernel<<<dim3(NT_ / 64, NB * NH), 128, 0, stream>>>(qbf, kbf, vtbf, yf);
  f2bf_kernel<<<EW_BLOCKS, 256, 0, stream>>>(yf, ybf);

  // 6) x1 = x + y @ proj_w + proj_b
  gemm_bf16_kernel<EPI_ADD_RESID><<<dim3(ND / 128, NTOK / 64), 256, 0, stream>>>(
      ybf, projwt, proj_b, x1, nullptr, x, nullptr, 0, NTOK, ND, ND);

  // 7) LN2 -> z (fp32 + bf16); router top-2 gates
  ln_kernel<<<NTOK, 256, 0, stream>>>(x1, ln2_w, ln2_b, zf, zbf);
  router_gate_kernel<<<NTOK / 8, 256, 0, stream>>>(zf, rw, gate);

  // 8) out = x1 + z  (double residual), then accumulate gated experts
  add2_kernel<<<EW_BLOCKS, 256, 0, stream>>>(x1, zf, out);

  for (int e = 0; e < NE; ++e) {
    gemm_bf16_kernel<EPI_GELU_BF16><<<dim3(NDFF / 128, NTOK / 64), 256, 0, stream>>>(
        zbf, w1t + (size_t)e * ND * NDFF, b1 + (size_t)e * NDFF,
        nullptr, t1bf, nullptr, nullptr, 0, NTOK, NDFF, ND);
    gemm_bf16_kernel<EPI_MOE_ACC><<<dim3(ND / 128, NTOK / 64), 256, 0, stream>>>(
        t1bf, w2t + (size_t)e * NDFF * ND, b2 + (size_t)e * ND,
        out, nullptr, nullptr, gate, e, NTOK, ND, NDFF);
  }
}
